// ResidualTokenizer_91336774516886
// MI455X (gfx1250) — compile-verified, matrix operand order checked by reading, not run on hardware
//
#include <hip/hip_runtime.h>
#include <hip/hip_bf16.h>
#include <math.h>

// ---- CDNA5 WMMA types -------------------------------------------------------
typedef __bf16 v16bf __attribute__((ext_vector_type(16)));
typedef float  v8f   __attribute__((ext_vector_type(8)));

#define NTOK   32768      // 32*16*64 tokens
#define NCODE  1024
#define NQ     8

__device__ __forceinline__ __bf16 f2bf(float f) {
  unsigned u = __builtin_bit_cast(unsigned, f);
  u += 0x7FFFu + ((u >> 16) & 1u);          // round-to-nearest-even
  unsigned short h = (unsigned short)(u >> 16);
  return __builtin_bit_cast(__bf16, h);
}

// slot s (0..15) of a v16bf A/B fragment -> k index within the 32-wide K window
// (ISA 7.12.2: 16-bit A 16x32: lanes0-15 K={0..7,16..23}, lanes16-31 K={8..15,24..31},
//  2 values per VGPR)
__device__ __forceinline__ int klocal(int s, int hh) {
  int j = s >> 1, p = s & 1;
  return ((j >> 2) << 4) + (hh << 3) + ((j & 3) << 1) + p;
}

#define WMMA_BF16(A, Bv, C) \
  __builtin_amdgcn_wmma_f32_16x16x32_bf16(false, (A), false, (Bv), (short)0, (C), false, false)

// ---- K1: 64-pt DFT magnitude per patch -------------------------------------
// x: [32,4096,16] f32 ; zor: [NTOK,64] ; patch index = ((b*16+v)*64+l)
__global__ void k_fft(const float* __restrict__ x, float* __restrict__ zor) {
  __shared__ float ct[64], st[64], zp[4][64];
  int tid = threadIdx.x;
  if (tid < 64) {
    float ang = (float)tid * (6.2831853071795864f / 64.0f);
    ct[tid] = cosf(ang);
    st[tid] = sinf(ang);
  }
  int g = tid >> 6, k = tid & 63;
  int patch = blockIdx.x * 4 + g;
  int b = patch >> 10, v = (patch >> 6) & 15, l = patch & 63;
  zp[g][k] = x[((b * 4096 + l * 64 + k) << 4) + v];
  __syncthreads();
  float re = 0.f, im = 0.f;
  #pragma unroll 8
  for (int d = 0; d < 64; ++d) {
    float zv = zp[g][d];
    int j = (k * d) & 63;
    re += zv * ct[j];
    im += zv * st[j];
  }
  zor[patch * 64 + k] = sqrtf(re * re + im * im);
}

// ---- K2: per-l mean / unbiased std -----------------------------------------
__global__ void k_stats(const float* __restrict__ zor, float* meanb, float* stdeps) {
  int l = blockIdx.x, tid = threadIdx.x;
  float s = 0.f, s2 = 0.f;
  for (int i = tid; i < 32768; i += 256) {
    int g = i >> 6, d = i & 63;
    float v = zor[((g << 6) + l) * 64 + d];
    s += v; s2 += v * v;
  }
  __shared__ float sh[256], sh2[256];
  sh[tid] = s; sh2[tid] = s2; __syncthreads();
  for (int o = 128; o > 0; o >>= 1) {
    if (tid < o) { sh[tid] += sh[tid + o]; sh2[tid] += sh2[tid + o]; }
    __syncthreads();
  }
  if (tid == 0) {
    float m = sh[0] / 32768.f;
    float var = (sh2[0] - 32768.f * m * m) / 32767.f;
    meanb[l] = m;
    stdeps[l] = sqrtf(fmaxf(var, 0.f)) + 1e-5f;
  }
}

// ---- K3: fused normalize + encoder GEMM (WMMA bf16) ------------------------
__global__ void k_enc(const float* __restrict__ zor, const float* __restrict__ meanb,
                      const float* __restrict__ stdeps, const float* __restrict__ enc_w,
                      const float* __restrict__ enc_b, float* __restrict__ zh) {
  int wave = threadIdx.x >> 5, lane = threadIdx.x & 31;
  int m = lane & 15, hh = lane >> 4;
  int row0 = (blockIdx.x * 8 + wave) * 16;
  int row = row0 + m;
  int l = row & 63;
  float mu = meanb[l], inv = 1.0f / stdeps[l];
  v16bf a0, a1;
  #pragma unroll
  for (int s = 0; s < 16; ++s) {
    int k0 = klocal(s, hh);
    a0[s] = f2bf((zor[row * 64 + k0]      - mu) * inv);
    a1[s] = f2bf((zor[row * 64 + 32 + k0] - mu) * inv);
  }
  for (int ct = 0; ct < 4; ++ct) {
    int e = ct * 16 + m;
    v16bf b0, b1;
    #pragma unroll
    for (int s = 0; s < 16; ++s) {
      int k0 = klocal(s, hh);
      b0[s] = f2bf(enc_w[e * 64 + k0]);
      b1[s] = f2bf(enc_w[e * 64 + 32 + k0]);
    }
    v8f c = {};
    c = WMMA_BF16(a0, b0, c);
    c = WMMA_BF16(a1, b1, c);
    float bias = enc_b[e];
    #pragma unroll
    for (int i = 0; i < 8; ++i) {
      int r = row0 + i + 8 * hh;
      zh[r * 64 + e] = c[i] + bias;
    }
  }
}

// ---- K4: per-(b,e) L2 norm over 1024 tokens --------------------------------
__global__ void k_norm(const float* __restrict__ zh, float* __restrict__ invn) {
  int b = blockIdx.x, tid = threadIdx.x;
  int e = tid & 63, g = tid >> 6;
  float s = 0.f;
  for (int n = g; n < 1024; n += 4) {
    float v = zh[(b * 1024 + n) * 64 + e];
    s += v * v;
  }
  __shared__ float sh[256];
  sh[tid] = s; __syncthreads();
  if (g == 0) {
    float t = sh[e] + sh[64 + e] + sh[128 + e] + sh[192 + e];
    invn[(b << 6) + e] = 1.0f / sqrtf(t);
  }
}

// ---- K5: prep: 0.5*||c||^2 + bf16-swizzled codebook + zero loss ------------
__global__ void k_prep(const float* __restrict__ cbs, float* __restrict__ hc2,
                       __bf16* __restrict__ cbswz, float* __restrict__ lossS) {
  int gid = blockIdx.x * 256 + threadIdx.x;   // < 32768
  if (gid < 8) lossS[gid] = 0.f;
  if (gid < 8192) {
    int q = gid >> 10, c = gid & 1023;
    const float* rowp = cbs + ((size_t)q * 1024 + c) * 64;
    float s = 0.f;
    #pragma unroll 8
    for (int d = 0; d < 64; ++d) { float v = rowp[d]; s += v * v; }
    hc2[gid] = 0.5f * s;
  }
  int lane = gid & 31, t = (gid >> 5) & 1, c = (gid >> 6) & 63, q = gid >> 12;
  int n = lane & 15, hh = lane >> 4;
  v16bf frag;
  #pragma unroll
  for (int s = 0; s < 16; ++s) {
    int k = t * 32 + klocal(s, hh);
    frag[s] = f2bf(cbs[((size_t)q * 1024 + (c * 16 + n)) * 64 + k]);
  }
  ((v16bf*)cbswz)[gid] = frag;
}

// ---- K6: scale tokens, init residual & quantized_out -----------------------
__global__ void k_init(const float* __restrict__ zh, const float* __restrict__ invn,
                       float* __restrict__ res, float* __restrict__ qout) {
  int gid = blockIdx.x * 256 + threadIdx.x;
  int row = gid >> 6, e = gid & 63, b = row >> 10;
  res[gid] = zh[gid] * invn[(b << 6) + e];
  qout[gid] = 0.f;
}

// ---- per-tile argmax reduce + gather + residual update ---------------------
__device__ __forceinline__ float finalize_tile(int row0, int lane, int hh,
                                               float best[8], int bidx[8],
                                               float* __restrict__ res,
                                               float* __restrict__ qout,
                                               const float* __restrict__ cb,
                                               float* __restrict__ codes_out) {
  // reduce over the 16 lanes sharing each row (xor butterfly stays in half-wave)
  #pragma unroll
  for (int i = 0; i < 8; ++i) {
    #pragma unroll
    for (int off = 1; off < 16; off <<= 1) {
      float ov = __shfl_xor(best[i], off);
      int   oi = __shfl_xor(bidx[i], off);
      if (ov > best[i] || (ov == best[i] && oi < bidx[i])) { best[i] = ov; bidx[i] = oi; }
    }
  }
  // exchange halves so every lane has indices for rows 0..15
  int idxLo[8], idxHi[8];
  #pragma unroll
  for (int i = 0; i < 8; ++i) {
    int oi = __shfl_xor(bidx[i], 16);
    idxLo[i] = hh ? oi : bidx[i];
    idxHi[i] = hh ? bidx[i] : oi;
  }
  if (lane == 0) {
    #pragma unroll
    for (int i = 0; i < 8; ++i) {
      codes_out[row0 + i]     = (float)idxLo[i];
      codes_out[row0 + 8 + i] = (float)idxHi[i];
    }
  }
  // gather + residual update + loss, exact f32 (tile owned by one wave)
  float lossAcc = 0.f;
  #pragma unroll
  for (int i = 0; i < 16; ++i) {
    int ridx = (i < 8) ? idxLo[i] : idxHi[i - 8];
    int r = row0 + i;
    #pragma unroll
    for (int dd = 0; dd < 2; ++dd) {
      int d = lane + dd * 32;
      float qv   = cb[ridx * 64 + d];
      float rold = res[r * 64 + d];
      float rn   = rold - qv;
      res[r * 64 + d]  = rn;
      qout[r * 64 + d] += qv;
      lossAcc += rn * rn;
    }
  }
  return lossAcc;
}

// ---- K7: one RVQ stage ------------------------------------------------------
// Each wave owns TWO 16-row tiles and shares the B fragments between them:
// 4 WMMAs per 2 v16bf loads -> half the L2 traffic, double WMMA density.
__global__ void k_stage(float* __restrict__ res, float* __restrict__ qout,
                        const float* __restrict__ cb,        // f32 [1024][64] this stage
                        const __bf16* __restrict__ cbswz,    // swizzled bf16, this stage
                        const float* __restrict__ hc2,       // 0.5*||c||^2, this stage
                        float* __restrict__ codes_out,       // d_out slice, this stage
                        float* __restrict__ lossSum) {
  int wave = threadIdx.x >> 5, lane = threadIdx.x & 31;
  int m = lane & 15, hh = lane >> 4;
  const v16bf* bs = (const v16bf*)cbswz;

  int tileA = (blockIdx.x * 8 + wave) * 2;    // 128 blocks * 8 waves * 2 = 2048 tiles
  int rowA0 = tileA * 16, rowB0 = rowA0 + 16;
  int rowA = rowA0 + m,  rowB = rowB0 + m;

  v16bf a0, a1, a2, a3;
  #pragma unroll
  for (int s = 0; s < 16; ++s) {
    int k0 = klocal(s, hh);
    a0[s] = f2bf(res[rowA * 64 + k0]);
    a1[s] = f2bf(res[rowA * 64 + 32 + k0]);
    a2[s] = f2bf(res[rowB * 64 + k0]);
    a3[s] = f2bf(res[rowB * 64 + 32 + k0]);
  }
  float bestA[8], bestB[8]; int bidxA[8], bidxB[8];
  #pragma unroll
  for (int i = 0; i < 8; ++i) {
    bestA[i] = -3.4e38f; bidxA[i] = 0;
    bestB[i] = -3.4e38f; bidxB[i] = 0;
  }

  for (int ct = 0; ct < 64; ++ct) {           // 1024 codes / 16
    v16bf b0 = bs[(ct * 2 + 0) * 32 + lane];
    v16bf b1 = bs[(ct * 2 + 1) * 32 + lane];
    v8f cA = {}, cB = {};
    cA = WMMA_BF16(a0, b0, cA);
    cB = WMMA_BF16(a2, b0, cB);
    cA = WMMA_BF16(a1, b1, cA);
    cB = WMMA_BF16(a3, b1, cB);
    int code = ct * 16 + m;
    float h = hc2[code];
    #pragma unroll
    for (int i = 0; i < 8; ++i) {
      float scA = cA[i] - h;                   // r.c - 0.5||c||^2 (argmax == argmin d2)
      if (scA > bestA[i]) { bestA[i] = scA; bidxA[i] = code; }
      float scB = cB[i] - h;
      if (scB > bestB[i]) { bestB[i] = scB; bidxB[i] = code; }
    }
  }

  float lossAcc = 0.f;
  lossAcc += finalize_tile(rowA0, lane, hh, bestA, bidxA, res, qout, cb, codes_out);
  lossAcc += finalize_tile(rowB0, lane, hh, bestB, bidxB, res, qout, cb, codes_out);

  #pragma unroll
  for (int off = 1; off < 32; off <<= 1) lossAcc += __shfl_xor(lossAcc, off);
  if (lane == 0) atomicAdd(lossSum, lossAcc);
}

// ---- K8: decoder GEMM + de-normalize (WMMA bf16) ---------------------------
__global__ void k_dec(const float* __restrict__ qout, const float* __restrict__ dec_w,
                      const float* __restrict__ dec_b, const float* __restrict__ meanb,
                      const float* __restrict__ stdeps, float* __restrict__ out) {
  int wave = threadIdx.x >> 5, lane = threadIdx.x & 31;
  int m = lane & 15, hh = lane >> 4;
  int row0 = (blockIdx.x * 8 + wave) * 16;
  int row = row0 + m;
  v16bf a0, a1;
  #pragma unroll
  for (int s = 0; s < 16; ++s) {
    int k0 = klocal(s, hh);
    a0[s] = f2bf(qout[row * 64 + k0]);
    a1[s] = f2bf(qout[row * 64 + 32 + k0]);
  }
  for (int ct = 0; ct < 4; ++ct) {
    int e = ct * 16 + m;
    v16bf b0, b1;
    #pragma unroll
    for (int s = 0; s < 16; ++s) {
      int k0 = klocal(s, hh);
      b0[s] = f2bf(dec_w[e * 64 + k0]);
      b1[s] = f2bf(dec_w[e * 64 + 32 + k0]);
    }
    v8f c = {};
    c = WMMA_BF16(a0, b0, c);
    c = WMMA_BF16(a1, b1, c);
    float bias = dec_b[e];
    #pragma unroll
    for (int i = 0; i < 8; ++i) {
      int r = row0 + i + 8 * hh;
      int l = r & 63;
      out[r * 64 + e] = (c[i] + bias) * stdeps[l] + meanb[l];
    }
  }
}

// ---- K9: final commit loss --------------------------------------------------
__global__ void k_loss(const float* __restrict__ lossS, float* __restrict__ outLoss) {
  if (threadIdx.x == 0) {
    float s = 0.f;
    for (int q = 0; q < NQ; ++q) s += lossS[q];
    outLoss[0] = s / (32768.f * 64.f) / 8.f;
  }
}

// ---- host launch ------------------------------------------------------------
extern "C" void kernel_launch(void* const* d_in, const int* in_sizes, int n_in,
                              void* d_out, int out_size, void* d_ws, size_t ws_size,
                              hipStream_t stream) {
  const float* x     = (const float*)d_in[0];
  const float* enc_w = (const float*)d_in[1];
  const float* enc_b = (const float*)d_in[2];
  const float* dec_w = (const float*)d_in[3];
  const float* dec_b = (const float*)d_in[4];
  const float* cbs   = (const float*)d_in[5];
  float* out = (float*)d_out;

  float* w      = (float*)d_ws;           // ~26.3 MB total
  float* zor    = w;                      // [NTOK*64], reused as qout after K3
  float* zh     = w + 2097152;            // [NTOK*64]
  float* res    = w + 4194304;            // [NTOK*64]
  float* meanb  = w + 6291456;            // [64]
  float* stdeps = w + 6291520;            // [64]
  float* invn   = w + 6291584;            // [2048]
  float* hc2    = w + 6293632;            // [8192]
  float* lossS  = w + 6301824;            // [8]
  __bf16* cbswz = (__bf16*)(w + 6301856); // [524288] bf16 (32B-aligned)
  float* qout = zor;

  k_fft  <<<8192, 256, 0, stream>>>(x, zor);
  k_prep <<<128,  256, 0, stream>>>(cbs, hc2, cbswz, lossS);
  k_stats<<<64,   256, 0, stream>>>(zor, meanb, stdeps);
  k_enc  <<<256,  256, 0, stream>>>(zor, meanb, stdeps, enc_w, enc_b, zh);
  k_norm <<<32,   256, 0, stream>>>(zh, invn);
  k_init <<<8192, 256, 0, stream>>>(zh, invn, res, qout);
  for (int q = 0; q < NQ; ++q) {
    k_stage<<<128, 256, 0, stream>>>(res, qout,
                                     cbs + (size_t)q * NCODE * 64,
                                     cbswz + (size_t)q * 65536,
                                     hc2 + q * NCODE,
                                     out + 2097152 + (size_t)q * NTOK,
                                     lossS + q);
  }
  k_dec  <<<256, 256, 0, stream>>>(qout, dec_w, dec_b, meanb, stdeps, out);
  k_loss <<<1, 32, 0, stream>>>(lossS, out + 2359296);
}
